// CycleMultiBlock_82678120448389
// MI455X (gfx1250) — compile-verified, matrix operand order checked by reading, not run on hardware
//
#include <hip/hip_runtime.h>
#include <stdint.h>

// ---------------------------------------------------------------------------
// CycleMultiBlock forward for MI455X (gfx1250, wave32, WMMA).
// B=16, S=2048, M=128, D=1024, L=4.  All GEMMs run as bf16 WMMA
// (v_wmma_f32_16x16x32_bf16) with f32 accumulation.  Global->LDS tile
// movement uses CDNA5 async loads (global_load_async_to_lds_b128, ASYNCcnt),
// no VGPR staging.  Each wave owns a 64x64 output tile (4x4 WMMA frags,
// 16 WMMAs per K-step) to halve LDS fragment traffic per WMMA vs 4x2.
// The rotation "scan" is a constant-angle recurrence (theta has no S
// dependence), one chain per lane with coalesced float2 streams.
// ---------------------------------------------------------------------------

#define DIM   1024
#define BBAT  16
#define SEQ   2048
#define VOCAB 128
#define NL    4
#define TOK   (BBAT * SEQ)          // 32768 tokens

typedef __attribute__((ext_vector_type(16))) __bf16 v16bf;
typedef __attribute__((ext_vector_type(8)))  float  v8f;

__device__ __forceinline__ unsigned short f2bf(float f) {
  uint32_t u = __float_as_uint(f);
  u += 0x7FFFu + ((u >> 16) & 1u);          // round-to-nearest-even
  return (unsigned short)(u >> 16);
}

// CDNA5 async global->LDS copy, 16B granule. LDS dest comes from a VGPR
// holding the LDS byte offset; completion tracked by ASYNCcnt.
__device__ __forceinline__ void async_ld_b128(uint32_t lds_off, const void* g) {
  asm volatile("global_load_async_to_lds_b128 %0, %1, off"
               :: "v"(lds_off), "v"(g) : "memory");
}
__device__ __forceinline__ uint32_t lds_off_of(const void* p) {
  return (uint32_t)(uintptr_t)p;   // generic-from-LDS: low 32 bits = LDS offset
}

// ---------------------------------------------------------------------------
// Weight convert + transpose:  W f32 [K][N]  ->  Wt bf16 [N][K]
// ---------------------------------------------------------------------------
__global__ __launch_bounds__(256) void convT_kernel(
    const float* __restrict__ W, unsigned short* __restrict__ Wt, int K, int N) {
  long gid = (long)blockIdx.x * 256 + threadIdx.x;   // over N*K
  long n = gid / K, k = gid - n * K;
  Wt[gid] = f2bf(W[k * (long)N + n]);
}

// ---------------------------------------------------------------------------
// Embedding gather: a[token][d] = emb[idx[token]][d]   (float4 granularity)
// ---------------------------------------------------------------------------
__global__ __launch_bounds__(256) void embed_kernel(
    const int* __restrict__ idx, const float* __restrict__ emb,
    float* __restrict__ A) {
  long gid = (long)blockIdx.x * 256 + threadIdx.x;   // over TOK*DIM/4
  long token = gid / (DIM / 4);
  long q = gid - token * (DIM / 4);
  int e = idx[token];
  ((float4*)A)[gid] = ((const float4*)(emb + (long)e * DIM))[q];
}

// ---------------------------------------------------------------------------
// LayerNorm (one wave per row of D=1024) -> bf16 output for GEMM A operand.
// ---------------------------------------------------------------------------
__global__ __launch_bounds__(256) void ln_kernel(
    const float* __restrict__ X, const float* __restrict__ g,
    const float* __restrict__ b, unsigned short* __restrict__ Y) {
  int wave = threadIdx.x >> 5, lane = threadIdx.x & 31;
  long row = (long)blockIdx.x * 8 + wave;
  const float* x = X + row * DIM;
  float4 v[8];
  float s = 0.f, s2 = 0.f;
#pragma unroll
  for (int c = 0; c < 8; ++c) {
    v[c] = *(const float4*)(x + c * 128 + lane * 4);
    s  += v[c].x + v[c].y + v[c].z + v[c].w;
    s2 += v[c].x * v[c].x + v[c].y * v[c].y + v[c].z * v[c].z + v[c].w * v[c].w;
  }
#pragma unroll
  for (int off = 16; off; off >>= 1) {
    s  += __shfl_xor(s, off, 32);
    s2 += __shfl_xor(s2, off, 32);
  }
  float mu  = s * (1.0f / DIM);
  float var = s2 * (1.0f / DIM) - mu * mu;
  float inv = rsqrtf(var + 1e-5f);
#pragma unroll
  for (int c = 0; c < 8; ++c) {
    int d = c * 128 + lane * 4;
    float4 gv = *(const float4*)(g + d);
    float4 bv = *(const float4*)(b + d);
    float o0 = (v[c].x - mu) * inv * gv.x + bv.x;
    float o1 = (v[c].y - mu) * inv * gv.y + bv.y;
    float o2 = (v[c].z - mu) * inv * gv.z + bv.z;
    float o3 = (v[c].w - mu) * inv * gv.w + bv.w;
    uint2 pk;
    pk.x = (uint32_t)f2bf(o0) | ((uint32_t)f2bf(o1) << 16);
    pk.y = (uint32_t)f2bf(o2) | ((uint32_t)f2bf(o3) << 16);
    *(uint2*)(Y + row * DIM + d) = pk;
  }
}

// ---------------------------------------------------------------------------
// Rotation recurrence + gate. theta constant along S so the associative scan
// is exactly h_t = R(theta) h_{t-1} + u_t.  One chain (batch b, block j) per
// lane; 8192 chains.  Output hv = h * v as packed bf16 pairs.
// ---------------------------------------------------------------------------
__global__ __launch_bounds__(256) void scan_kernel(
    const float* __restrict__ U, const float* __restrict__ V,
    const float* __restrict__ theta, unsigned short* __restrict__ HV) {
  int chain = blockIdx.x * 256 + threadIdx.x;        // 0 .. B*(D/2)-1
  int bb = chain >> 9;                               // / (D/2)
  int j  = chain & (DIM / 2 - 1);
  float th = theta[j];
  float c = __cosf(th), s = __sinf(th);
  float hx = 0.f, hy = 0.f;
  const long base = (long)bb * SEQ * (DIM / 2) + j;
  const float2* u2 = (const float2*)U + base;
  const float2* v2 = (const float2*)V + base;
  uint32_t*     o  = (uint32_t*)HV + base;
  for (int t = 0; t < SEQ; ++t) {
    float2 u = u2[(long)t * (DIM / 2)];
    float nx = c * hx - s * hy + u.x;
    float ny = s * hx + c * hy + u.y;
    hx = nx; hy = ny;
    float2 vv = v2[(long)t * (DIM / 2)];
    o[(long)t * (DIM / 2)] =
        (uint32_t)f2bf(hx * vv.x) | ((uint32_t)f2bf(hy * vv.y) << 16);
  }
}

// ---------------------------------------------------------------------------
// bf16 WMMA GEMM:  Out[rows,N] = A[rows,K] @ Bt[N,K]^T + bias (+R) (relu)
// Block tile 128x128x32, 128 threads = 4 waves in a 2x2 grid; each wave owns
// a 64x64 sub-tile = 4x4 WMMA 16x16 accumulators (16 WMMAs / 16 ds_load_b128
// per K-step).  Double-buffered LDS filled by async global->LDS b128
// (ASYNCcnt pipelined), padded stride (40 shorts) => conflict-free fragment
// reads and 16B-aligned async stores.
// MODE: 0=f32 out, 1=f32 out + residual, 2=relu -> bf16 out.
// ---------------------------------------------------------------------------
constexpr int BM = 128, BN = 128, BK = 32;
constexpr int LSTR = 40;   // shorts per LDS row (80B: 16B-aligned, no bank conflicts)

union FragU { v16bf v; uint32_t u[8]; };

__device__ __forceinline__ v16bf load_frag(const unsigned short* base, int row0,
                                           int lane) {
  FragU f;
  int m = lane & 15, half = lane >> 4;
  const unsigned short* p = base + (row0 + m) * LSTR;
#pragma unroll
  for (int v = 0; v < 8; ++v) {
    // 16-bit A/B fragment K pattern (ISA 7.12.2): VGPR v holds K pair
    int k = ((v < 4) ? 2 * v : 16 + 2 * (v - 4)) + half * 8;
    f.u[v] = *(const uint32_t*)(p + k);
  }
  return f.v;
}

template <int MODE>
__global__ __launch_bounds__(128) void gemm_bf16_kernel(
    const unsigned short* __restrict__ A,   // [rows,K] bf16
    const unsigned short* __restrict__ Bt,  // [N,K]    bf16 (B transposed)
    const float* __restrict__ bias,         // [N]
    const float* __restrict__ R,            // residual [rows,N] (MODE 1)
    void* __restrict__ Out,                 // f32 (MODE 0/1) or bf16 (MODE 2)
    int K, int N) {
  __shared__ unsigned short sA[2][BM * LSTR];
  __shared__ unsigned short sB[2][BN * LSTR];

  const int tid = threadIdx.x;              // 0..127
  const int lane = tid & 31, wave = tid >> 5;
  const int wr = wave >> 1, wc = wave & 1;  // 2x2 wave grid, 64x64 per wave
  const long rowBase = (long)blockIdx.x * BM;
  const int  colBase = blockIdx.y * BN;

  // Per-thread async-copy chunks: each tile = 512 16B chunks; 128 threads ->
  // 4 chunks of A + 4 chunks of B per thread per tile pair (8 async ops).
  const int r0 = tid >> 2, cc0 = (tid & 3) * 8;   // chunk tid (rows 0..31)
  const unsigned short* Ag[4];
  const unsigned short* Bg[4];
  uint32_t la[4], lb[4];
#pragma unroll
  for (int p = 0; p < 4; ++p) {
    int r = r0 + 32 * p;
    Ag[p] = A  + (rowBase + r) * K + cc0;
    Bg[p] = Bt + (long)(colBase + r) * K + cc0;
    la[p] = lds_off_of(&sA[0][r * LSTR + cc0]);
    lb[p] = lds_off_of(&sB[0][r * LSTR + cc0]);
  }
  const uint32_t bufB = (uint32_t)(BM * LSTR * 2);  // bytes sA[0] -> sA[1]

  auto issue = [&](int kb) {
    const uint32_t d = (kb & 1) ? bufB : 0u;
    const long o = (long)kb * BK;                   // shorts
#pragma unroll
    for (int p = 0; p < 4; ++p) async_ld_b128(la[p] + d, Ag[p] + o);
#pragma unroll
    for (int p = 0; p < 4; ++p) async_ld_b128(lb[p] + d, Bg[p] + o);
  };

  v8f acc[4][4];
  v8f zero = {};
#pragma unroll
  for (int i = 0; i < 4; ++i)
#pragma unroll
    for (int j = 0; j < 4; ++j) acc[i][j] = zero;

  issue(0);

  const int KT = K / BK;
  for (int kb = 0; kb < KT; ++kb) {
    const int buf = kb & 1;
    if (kb + 1 < KT) {
      issue(kb + 1);  // into the other buffer (free since end of kb-1)
      // tile kb's 8 ops (issued first, in-order completion) are done when
      // outstanding count drops to the 8 just issued:
      asm volatile("s_wait_asynccnt 0x8" ::: "memory");
    } else {
      asm volatile("s_wait_asynccnt 0x0" ::: "memory");
    }
    __syncthreads();   // everyone's tile-kb data visible in LDS

    v16bf af[4], bfr[4];
#pragma unroll
    for (int i = 0; i < 4; ++i)
      af[i] = load_frag(sA[buf], wr * 64 + i * 16, lane);
#pragma unroll
    for (int j = 0; j < 4; ++j)
      bfr[j] = load_frag(sB[buf], wc * 64 + j * 16, lane);

#pragma unroll
    for (int i = 0; i < 4; ++i)
#pragma unroll
      for (int j = 0; j < 4; ++j)
        acc[i][j] = __builtin_amdgcn_wmma_f32_16x16x32_bf16(
            false, af[i], false, bfr[j], (short)0, acc[i][j], false, false);

    __syncthreads();   // all reads of buf done before it is refilled (kb+2)
  }

  // Epilogue. C/D layout: lane n = lane&15, VGPR r -> m = r + 8*(lane>>4).
  const int half = lane >> 4, nn = lane & 15;
#pragma unroll
  for (int i = 0; i < 4; ++i) {
#pragma unroll
    for (int j = 0; j < 4; ++j) {
#pragma unroll
      for (int r = 0; r < 8; ++r) {
        long gm = rowBase + wr * 64 + i * 16 + r + 8 * half;
        int  gn = colBase + wc * 64 + j * 16 + nn;
        float val = acc[i][j][r] + bias[gn];
        if (MODE == 1) val += R[gm * N + gn];
        if (MODE == 2) {
          val = val > 0.f ? val : 0.f;
          ((unsigned short*)Out)[gm * N + gn] = f2bf(val);
        } else {
          ((float*)Out)[gm * N + gn] = val;
        }
      }
    }
  }
}

// ---------------------------------------------------------------------------
// Launcher
// ---------------------------------------------------------------------------
extern "C" void kernel_launch(void* const* d_in, const int* in_sizes, int n_in,
                              void* d_out, int out_size, void* d_ws,
                              size_t ws_size, hipStream_t stream) {
  (void)in_sizes; (void)n_in; (void)out_size; (void)ws_size;
  const int*   input_x = (const int*)d_in[0];
  const float* emb   = (const float*)d_in[1];
  const float* theta = (const float*)d_in[2];
  const float* Wx = (const float*)d_in[3];
  const float* bx = (const float*)d_in[4];
  const float* Wv = (const float*)d_in[5];
  const float* bv = (const float*)d_in[6];
  const float* Wh = (const float*)d_in[7];
  const float* bh = (const float*)d_in[8];
  const float* ln_g = (const float*)d_in[9];
  const float* ln_b = (const float*)d_in[10];
  const float* lnf_g = (const float*)d_in[11];
  const float* lnf_b = (const float*)d_in[12];
  const float* Wo1 = (const float*)d_in[13];
  const float* bo1 = (const float*)d_in[14];
  const float* Wo2 = (const float*)d_in[15];
  const float* bo2 = (const float*)d_in[16];
  float* out = (float*)d_out;

  // Workspace carve-out (256B aligned slices)
  char* ws = (char*)d_ws;
  size_t off = 0;
  auto carve = [&](size_t bytes) -> char* {
    char* p = ws + off;
    off += (bytes + 255) & ~(size_t)255;
    return p;
  };
  const size_t DD = (size_t)DIM * DIM;
  float*          a    = (float*)         carve((size_t)TOK * DIM * 4);
  unsigned short* xb   = (unsigned short*)carve((size_t)TOK * DIM * 2);
  float*          ub   = (float*)         carve((size_t)TOK * DIM * 4);
  float*          vb   = (float*)         carve((size_t)TOK * DIM * 4);
  unsigned short* hv   = (unsigned short*)carve((size_t)TOK * DIM * 2);
  unsigned short* WxT  = (unsigned short*)carve(NL * DD * 2);
  unsigned short* WvT  = (unsigned short*)carve(NL * DD * 2);
  unsigned short* WhT  = (unsigned short*)carve(NL * DD * 2);
  unsigned short* Wo1T = (unsigned short*)carve(DD * 2);
  unsigned short* Wo2T = (unsigned short*)carve((size_t)VOCAB * DIM * 2);

  const dim3 blk(256);
  const dim3 gblk(128);                       // GEMM: 4 waves per block
  const dim3 gemmGridD(TOK / BM, DIM / BN);   // (256, 8)
  const dim3 gemmGridM(TOK / BM, VOCAB / BN); // (256, 1)
  const int convBlocksDD = (int)(DD / 256);

  // 1) weight convert + transpose (weights then live in L2 as bf16)
  for (int l = 0; l < NL; ++l) {
    convT_kernel<<<convBlocksDD, blk, 0, stream>>>(Wx + l * DD, WxT + l * DD, DIM, DIM);
    convT_kernel<<<convBlocksDD, blk, 0, stream>>>(Wv + l * DD, WvT + l * DD, DIM, DIM);
    convT_kernel<<<convBlocksDD, blk, 0, stream>>>(Wh + l * DD, WhT + l * DD, DIM, DIM);
  }
  convT_kernel<<<convBlocksDD, blk, 0, stream>>>(Wo1, Wo1T, DIM, DIM);
  convT_kernel<<<(VOCAB * DIM) / 256, blk, 0, stream>>>(Wo2, Wo2T, DIM, VOCAB);

  // 2) embedding gather
  embed_kernel<<<(TOK * (DIM / 4)) / 256, blk, 0, stream>>>(input_x, emb, a);

  // 3) layers
  for (int l = 0; l < NL; ++l) {
    ln_kernel<<<TOK / 8, blk, 0, stream>>>(a, ln_g + l * DIM, ln_b + l * DIM, xb);
    gemm_bf16_kernel<0><<<gemmGridD, gblk, 0, stream>>>(xb, WxT + l * DD, bx + l * DIM, nullptr, ub, DIM, DIM);
    gemm_bf16_kernel<0><<<gemmGridD, gblk, 0, stream>>>(xb, WvT + l * DD, bv + l * DIM, nullptr, vb, DIM, DIM);
    scan_kernel<<<(BBAT * (DIM / 2)) / 256, blk, 0, stream>>>(ub, vb, theta + l * (DIM / 2), hv);
    if (l < NL - 1) {
      gemm_bf16_kernel<1><<<gemmGridD, gblk, 0, stream>>>(hv, WhT + l * DD, bh + l * DIM, a, a, DIM, DIM);
    } else {
      gemm_bf16_kernel<0><<<gemmGridD, gblk, 0, stream>>>(hv, WhT + l * DD, bh + l * DIM, nullptr, a, DIM, DIM);
    }
  }

  // 4) final LN + output head (relu GEMM -> bf16, then vocab GEMM -> f32 out)
  ln_kernel<<<TOK / 8, blk, 0, stream>>>(a, lnf_g, lnf_b, xb);
  gemm_bf16_kernel<2><<<gemmGridD, gblk, 0, stream>>>(xb, Wo1T, bo1, nullptr, hv, DIM, DIM);
  gemm_bf16_kernel<0><<<gemmGridM, gblk, 0, stream>>>(hv, Wo2T, bo2, nullptr, out, DIM, VOCAB);
}
